// ClassMLP_47330539602099
// MI455X (gfx1250) — matrix operand order repaired and checked by reading, not compiled, not run on hardware
//
#include <hip/hip_runtime.h>
#include <hip/hip_bf16.h>

// ---------------------------------------------------------------------------
// Problem constants (from reference)
// ---------------------------------------------------------------------------
#define BB      8192
#define LL      16
#define DMODEL  512
#define DINNER  1024
#define DCONV   4
#define DTRANK  32
#define HID     1024
#define OUTC    64
#define EPSF    1e-5f

typedef _Float16 h16 __attribute__((ext_vector_type(16)));
typedef _Float16 h8  __attribute__((ext_vector_type(8)));
typedef float    f8  __attribute__((ext_vector_type(8)));

// ---------------------------------------------------------------------------
// Fragment loaders (CDNA5 WMMA 16x16x32 f16 layouts, wave32)
//   A 16x32 (MxK): lane<16 holds row=lane, K {0..7,16..23}; lane>=16 K {8..15,24..31}
//   B 32x16 (KxN): lane holds col=lane&15, K same as A (lo half lanes K {0..7,16..23}, hi K {8..15,24..31})
//   C/D 16x16:     VGPR j -> row j (lanes<16) / row 8+j (lanes>=16); col = lane&15
// ---------------------------------------------------------------------------
__device__ __forceinline__ h16 load_a16(const _Float16* __restrict__ arow, int k0) {
  h8 c0 = *(const h8*)(arow + k0);
  h8 c1 = *(const h8*)(arow + k0 + 16);
  h16 a;
#pragma unroll
  for (int i = 0; i < 8; ++i) { a[i] = c0[i]; a[8 + i] = c1[i]; }
  return a;
}

__device__ __forceinline__ h16 load_b16(const _Float16* __restrict__ W, int ldw,
                                        int n, int k0, int half_id, int ncol) {
  const _Float16* p = W + (size_t)(n + ncol) * ldw + k0 + half_id * 8;
  h8 d0 = *(const h8*)p;
  h8 d1 = *(const h8*)(p + 16);
  h16 b;
#pragma unroll
  for (int i = 0; i < 8; ++i) { b[i] = d0[i]; b[8 + i] = d1[i]; }
  return b;
}

__device__ __forceinline__ f8 wmma_f16(h16 a, h16 b, f8 c) {
  return __builtin_amdgcn_wmma_f32_16x16x32_f16(false, a, false, b, (short)0, c,
                                                false, false);
}

__device__ __forceinline__ float siluf(float x) { return x / (1.f + expf(-x)); }

// ---------------------------------------------------------------------------
// fp32 -> fp16 conversion (x and weight staging)
// ---------------------------------------------------------------------------
__global__ void cvt_f32_f16(const float* __restrict__ s, _Float16* __restrict__ d,
                            long n) {
  long i = (long)blockIdx.x * blockDim.x + threadIdx.x;
  if (i < n) d[i] = (_Float16)s[i];
}

// ---------------------------------------------------------------------------
// K1: xz = x @ in_proj_w^T  fused with causal conv + SiLU.
// One wave per block, ONE BLOCK PER BATCH: the block's 16 A-rows (16 KB f16)
// are looped over all 16 N-groups of 128 columns, so x is read from HBM once
// (subsequent A re-reads hit L0/L2). Columns < 1024 -> conv path (write xs
// f32); columns >= 1024 -> z, keep only t=15 and store silu(z15).
// ---------------------------------------------------------------------------
__global__ __launch_bounds__(32) void inproj_conv_kernel(
    const _Float16* __restrict__ xh, const _Float16* __restrict__ Wh,
    const float* __restrict__ conv_w, const float* __restrict__ conv_b,
    float* __restrict__ xs, float* __restrict__ sz15) {
  constexpr int NT = 8;
  const int lane = threadIdx.x & 31;
  const int half_id = lane >> 4, mr = lane & 15;
  const int b = blockIdx.x;
  const int m0 = b * 16;

  __shared__ float tile[16][NT * 16 + 4];
  const _Float16* arow = xh + (size_t)(m0 + mr) * DMODEL + (half_id ? 8 : 0);

  for (int ng = 0; ng < (2 * DINNER) / (NT * 16); ++ng) {
    const int n0 = ng * (NT * 16);

    f8 acc[NT];
#pragma unroll
    for (int nt = 0; nt < NT; ++nt)
#pragma unroll
      for (int j = 0; j < 8; ++j) acc[nt][j] = 0.f;

    for (int k0 = 0; k0 < DMODEL; k0 += 32) {
      h16 a = load_a16(arow, k0);
#pragma unroll
      for (int nt = 0; nt < NT; ++nt) {
        h16 bf = load_b16(Wh, DMODEL, n0 + nt * 16, k0, half_id, mr);
        acc[nt] = wmma_f16(a, bf, acc[nt]);
      }
    }

    __syncthreads();  // previous group's tile reads complete
#pragma unroll
    for (int nt = 0; nt < NT; ++nt)
#pragma unroll
      for (int j = 0; j < 8; ++j)
        tile[j + 8 * half_id][nt * 16 + mr] = acc[nt][j];
    __syncthreads();

    if (n0 < DINNER) {
      for (int c = lane; c < NT * 16; c += 32) {
        int d = n0 + c;
        float w0 = conv_w[d * 4 + 0], w1 = conv_w[d * 4 + 1];
        float w2 = conv_w[d * 4 + 2], w3 = conv_w[d * 4 + 3];
        float cb = conv_b[d];
        float x0 = 0.f, x1 = 0.f, x2 = 0.f;
        size_t base = (size_t)b * 16 * DINNER + d;
#pragma unroll
        for (int t = 0; t < 16; ++t) {
          float x3 = tile[t][c];
          float s = cb + w0 * x0 + w1 * x1 + w2 * x2 + w3 * x3;
          xs[base + (size_t)t * DINNER] = siluf(s);
          x0 = x1; x1 = x2; x2 = x3;
        }
      }
    } else {
      for (int c = lane; c < NT * 16; c += 32) {
        int d = n0 - DINNER + c;
        sz15[(size_t)b * DINNER + d] = siluf(tile[15][c]);
      }
    }
  }
}

// ---------------------------------------------------------------------------
// K2: x_dbl = xs @ x_proj_w^T   (N = 34, padded to 48 with guards).
// A is f32 (scan needs full precision), converted in-register.
// ---------------------------------------------------------------------------
__global__ __launch_bounds__(32) void xproj_kernel(
    const float* __restrict__ xs, const _Float16* __restrict__ Wh,
    float* __restrict__ xdbl) {
  constexpr int NT = 3, NW = DTRANK + 2;  // 34 valid columns
  const int lane = threadIdx.x & 31;
  const int half_id = lane >> 4, mr = lane & 15;
  const int m0 = blockIdx.x * 16;

  f8 acc[NT];
#pragma unroll
  for (int nt = 0; nt < NT; ++nt)
#pragma unroll
    for (int j = 0; j < 8; ++j) acc[nt][j] = 0.f;

  const float* arow = xs + (size_t)(m0 + mr) * DINNER + (half_id ? 8 : 0);
  for (int k0 = 0; k0 < DINNER; k0 += 32) {
    float4 f0 = *(const float4*)(arow + k0);
    float4 f1 = *(const float4*)(arow + k0 + 4);
    float4 f2 = *(const float4*)(arow + k0 + 16);
    float4 f3 = *(const float4*)(arow + k0 + 20);
    h16 a;
    a[0] = (_Float16)f0.x;  a[1] = (_Float16)f0.y;
    a[2] = (_Float16)f0.z;  a[3] = (_Float16)f0.w;
    a[4] = (_Float16)f1.x;  a[5] = (_Float16)f1.y;
    a[6] = (_Float16)f1.z;  a[7] = (_Float16)f1.w;
    a[8] = (_Float16)f2.x;  a[9] = (_Float16)f2.y;
    a[10] = (_Float16)f2.z; a[11] = (_Float16)f2.w;
    a[12] = (_Float16)f3.x; a[13] = (_Float16)f3.y;
    a[14] = (_Float16)f3.z; a[15] = (_Float16)f3.w;
#pragma unroll
    for (int nt = 0; nt < NT; ++nt) {
      int n = nt * 16 + mr;
      h16 bf;
#pragma unroll
      for (int i = 0; i < 16; ++i) bf[i] = (_Float16)0;
      if (n < NW) bf = load_b16(Wh, DINNER, nt * 16, k0, half_id, mr);
      acc[nt] = wmma_f16(a, bf, acc[nt]);
    }
  }

#pragma unroll
  for (int nt = 0; nt < NT; ++nt) {
    int c = nt * 16 + mr;
    if (c < NW) {
#pragma unroll
      for (int j = 0; j < 8; ++j) {
        int row = m0 + j + 8 * half_id;
        xdbl[(size_t)row * NW + c] = acc[nt][j];
      }
    }
  }
}

// ---------------------------------------------------------------------------
// K3: fused dt_proj + softplus + selective scan (D_STATE = 1), emits
//     u[b,d] = (h15 * C15 + D_skip[d]*xs15) * silu(z15)   as f16.
// One block per batch; x_dbl rows for the batch staged in LDS.
// ---------------------------------------------------------------------------
__global__ __launch_bounds__(128) void scan_kernel(
    const float* __restrict__ xdbl, const float* __restrict__ xs,
    const float* __restrict__ sz15, const float* __restrict__ dtw,
    const float* __restrict__ dtb, const float* __restrict__ alog,
    const float* __restrict__ dskip, _Float16* __restrict__ u) {
  constexpr int NW = DTRANK + 2;
  __shared__ float sdbl[16 * NW];
  const int b = blockIdx.x;
  const int tid = threadIdx.x;
  for (int i = tid; i < 16 * NW; i += 128) sdbl[i] = xdbl[(size_t)b * 16 * NW + i];
  __syncthreads();

  for (int d = tid; d < DINNER; d += 128) {
    float w[DTRANK];
    const float4* wp = (const float4*)(dtw + (size_t)d * DTRANK);
#pragma unroll
    for (int q = 0; q < DTRANK / 4; ++q) {
      float4 v = wp[q];
      w[q * 4 + 0] = v.x; w[q * 4 + 1] = v.y;
      w[q * 4 + 2] = v.z; w[q * 4 + 3] = v.w;
    }
    float Ad = -expf(alog[d]);
    float db = dtb[d];
    float h = 0.f, xs_last = 0.f;
    const float* xsb = xs + (size_t)b * 16 * DINNER + d;
#pragma unroll
    for (int t = 0; t < 16; ++t) {
      const float* r = sdbl + t * NW;
      float dd = db;
#pragma unroll
      for (int j = 0; j < DTRANK; ++j) dd += r[j] * w[j];
      float delta = (dd > 20.f) ? dd : log1pf(expf(dd));
      float dA = expf(delta * Ad);
      float xst = xsb[(size_t)t * DINNER];
      h = dA * h + delta * r[DTRANK] * xst;  // Bc at column 32
      xs_last = xst;
    }
    float C15 = sdbl[15 * NW + DTRANK + 1];  // Cc at column 33
    float y = h * C15 + dskip[d] * xs_last;
    u[(size_t)b * DINNER + d] = (_Float16)(y * sz15[(size_t)b * DINNER + d]);
  }
}

// ---------------------------------------------------------------------------
// Generic WMMA GEMM  C[M,N] = A[M,K] @ W[N,K]^T  (+ optional bias/BN/ReLU).
// EPI==0: plain; EPI==1: relu(g*(acc+bias-m)*rsqrt(v+eps)+bb). OUTF16 selects
// f16 vs f32 output.
// ---------------------------------------------------------------------------
template <int NT, int EPI, int OUTF16>
__global__ __launch_bounds__(32) void gemm16(
    const _Float16* __restrict__ A, const _Float16* __restrict__ W,
    void* __restrict__ Cv, int K, int Nst, const float* __restrict__ bias,
    const float* __restrict__ g, const float* __restrict__ bb,
    const float* __restrict__ bm, const float* __restrict__ bv) {
  const int lane = threadIdx.x & 31;
  const int half_id = lane >> 4, mr = lane & 15;
  const int m0 = blockIdx.x * 16;
  const int n0 = blockIdx.y * (NT * 16);

  f8 acc[NT];
#pragma unroll
  for (int nt = 0; nt < NT; ++nt)
#pragma unroll
    for (int j = 0; j < 8; ++j) acc[nt][j] = 0.f;

  const _Float16* arow = A + (size_t)(m0 + mr) * K + (half_id ? 8 : 0);
  for (int k0 = 0; k0 < K; k0 += 32) {
    h16 a = load_a16(arow, k0);
#pragma unroll
    for (int nt = 0; nt < NT; ++nt) {
      h16 bf = load_b16(W, K, n0 + nt * 16, k0, half_id, mr);
      acc[nt] = wmma_f16(a, bf, acc[nt]);
    }
  }

#pragma unroll
  for (int nt = 0; nt < NT; ++nt) {
    int c = n0 + nt * 16 + mr;
    float scale = 1.f, shift = 0.f;
    if (EPI == 1) {
      float sc = g[c] * rsqrtf(bv[c] + EPSF);
      scale = sc;
      shift = (bias[c] - bm[c]) * sc + bb[c];
    }
#pragma unroll
    for (int j = 0; j < 8; ++j) {
      int row = m0 + j + 8 * half_id;
      float val = acc[nt][j];
      if (EPI == 1) val = fmaxf(val * scale + shift, 0.f);
      if (OUTF16)
        ((_Float16*)Cv)[(size_t)row * Nst + c] = (_Float16)val;
      else
        ((float*)Cv)[(size_t)row * Nst + c] = val;
    }
  }
}

// ---------------------------------------------------------------------------
// K7: logits = h2 @ lin2_w^T + lin2_b, then row-wise log_softmax over 64.
// One wave covers a full 16x64 tile -> softmax block-local via LDS.
// ---------------------------------------------------------------------------
__global__ __launch_bounds__(32) void head_kernel(
    const _Float16* __restrict__ A, const _Float16* __restrict__ W,
    const float* __restrict__ b2, float* __restrict__ out) {
  constexpr int NT = 4;
  const int lane = threadIdx.x & 31;
  const int half_id = lane >> 4, mr = lane & 15;
  const int m0 = blockIdx.x * 16;

  f8 acc[NT];
#pragma unroll
  for (int nt = 0; nt < NT; ++nt)
#pragma unroll
    for (int j = 0; j < 8; ++j) acc[nt][j] = 0.f;

  const _Float16* arow = A + (size_t)(m0 + mr) * HID + (half_id ? 8 : 0);
  for (int k0 = 0; k0 < HID; k0 += 32) {
    h16 a = load_a16(arow, k0);
#pragma unroll
    for (int nt = 0; nt < NT; ++nt) {
      h16 bf = load_b16(W, HID, nt * 16, k0, half_id, mr);
      acc[nt] = wmma_f16(a, bf, acc[nt]);
    }
  }

  __shared__ float tl[16][OUTC + 4];
#pragma unroll
  for (int nt = 0; nt < NT; ++nt) {
    int c = nt * 16 + mr;
    float bv = b2[c];
#pragma unroll
    for (int j = 0; j < 8; ++j) tl[j + 8 * half_id][c] = acc[nt][j] + bv;
  }
  __syncthreads();

  if (lane < 16) {
    int r = lane;
    float mx = tl[r][0];
#pragma unroll
    for (int c = 1; c < OUTC; ++c) mx = fmaxf(mx, tl[r][c]);
    float s = 0.f;
#pragma unroll
    for (int c = 0; c < OUTC; ++c) s += expf(tl[r][c] - mx);
    float lse = mx + logf(s);
#pragma unroll
    for (int c = 0; c < OUTC; ++c)
      out[(size_t)(m0 + r) * OUTC + c] = tl[r][c] - lse;
  }
}

// ---------------------------------------------------------------------------
// Host launcher
// ---------------------------------------------------------------------------
static inline char* align256(char* p) {
  return (char*)(((uintptr_t)p + 255) & ~(uintptr_t)255);
}

extern "C" void kernel_launch(void* const* d_in, const int* in_sizes, int n_in,
                              void* d_out, int out_size, void* d_ws,
                              size_t ws_size, hipStream_t stream) {
  const float* x        = (const float*)d_in[0];
  const float* in_w     = (const float*)d_in[1];
  const float* conv_w   = (const float*)d_in[2];
  const float* conv_b   = (const float*)d_in[3];
  const float* xproj_w  = (const float*)d_in[4];
  const float* dtp_w    = (const float*)d_in[5];
  const float* dtp_b    = (const float*)d_in[6];
  const float* A_log    = (const float*)d_in[7];
  const float* D_skip   = (const float*)d_in[8];
  const float* outp_w   = (const float*)d_in[9];
  const float* l0_w     = (const float*)d_in[10];
  const float* l0_b     = (const float*)d_in[11];
  const float* l1_w     = (const float*)d_in[12];
  const float* l1_b     = (const float*)d_in[13];
  const float* l2_w     = (const float*)d_in[14];
  const float* l2_b     = (const float*)d_in[15];
  const float* bn0_g    = (const float*)d_in[16];
  const float* bn0_b    = (const float*)d_in[17];
  const float* bn0_m    = (const float*)d_in[18];
  const float* bn0_v    = (const float*)d_in[19];
  const float* bn1_g    = (const float*)d_in[20];
  const float* bn1_b    = (const float*)d_in[21];
  const float* bn1_m    = (const float*)d_in[22];
  const float* bn1_v    = (const float*)d_in[23];

  // Workspace carve-up
  char* p = (char*)d_ws;
  auto grab = [&](size_t bytes) { char* r = p; p = align256(p + bytes); return r; };

  const long n_x   = (long)BB * LL * DMODEL;      // 67,108,864
  const long n_inw = 2L * DINNER * DMODEL;        // 1,048,576
  const long n_xpw = (long)(DTRANK + 2) * DINNER; // 34,816
  const long n_opw = (long)DMODEL * DINNER;       // 524,288
  const long n_l0w = (long)HID * DMODEL;          // 524,288
  const long n_l1w = (long)HID * HID;             // 1,048,576
  const long n_l2w = (long)OUTC * HID;            // 65,536

  _Float16* x_h    = (_Float16*)grab(n_x * 2);
  _Float16* inw_h  = (_Float16*)grab(n_inw * 2);
  _Float16* xpw_h  = (_Float16*)grab(n_xpw * 2);
  _Float16* opw_h  = (_Float16*)grab(n_opw * 2);
  _Float16* l0w_h  = (_Float16*)grab(n_l0w * 2);
  _Float16* l1w_h  = (_Float16*)grab(n_l1w * 2);
  _Float16* l2w_h  = (_Float16*)grab(n_l2w * 2);
  float*    xs     = (float*)grab((size_t)BB * LL * DINNER * 4);     // 512 MB
  float*    sz15   = (float*)grab((size_t)BB * DINNER * 4);          // 32 MB
  float*    xdbl   = (float*)grab((size_t)BB * LL * (DTRANK + 2) * 4);
  _Float16* u_h    = (_Float16*)grab((size_t)BB * DINNER * 2);
  _Float16* hl_h   = (_Float16*)grab((size_t)BB * DMODEL * 2);
  _Float16* h1_h   = (_Float16*)grab((size_t)BB * HID * 2);
  _Float16* h2_h   = (_Float16*)grab((size_t)BB * HID * 2);

  auto cvt = [&](const float* s, _Float16* d, long n) {
    cvt_f32_f16<<<(unsigned)((n + 255) / 256), 256, 0, stream>>>(s, d, n);
  };
  cvt(x, x_h, n_x);
  cvt(in_w, inw_h, n_inw);
  cvt(xproj_w, xpw_h, n_xpw);
  cvt(outp_w, opw_h, n_opw);
  cvt(l0_w, l0w_h, n_l0w);
  cvt(l1_w, l1w_h, n_l1w);
  cvt(l2_w, l2w_h, n_l2w);

  // 1) in_proj GEMM + causal conv + silu (+ silu(z15)); one block per batch,
  //    N-groups looped inside so x is read from HBM exactly once.
  inproj_conv_kernel<<<dim3(BB), 32, 0, stream>>>(x_h, inw_h, conv_w, conv_b,
                                                  xs, sz15);

  // 2) x_proj GEMM (N=34)
  xproj_kernel<<<dim3((BB * LL) / 16), 32, 0, stream>>>(xs, xpw_h, xdbl);

  // 3) dt_proj + softplus + selective scan -> u (f16)
  scan_kernel<<<dim3(BB), 128, 0, stream>>>(xdbl, xs, sz15, dtp_w, dtp_b, A_log,
                                            D_skip, u_h);

  // 4) out_proj (only t=15 rows exist in u): hlast = u @ out_proj_w^T
  gemm16<4, 0, 1><<<dim3(BB / 16, DMODEL / 64), 32, 0, stream>>>(
      u_h, opw_h, hl_h, DINNER, DMODEL, nullptr, nullptr, nullptr, nullptr,
      nullptr);

  // 5) lin0 + bn0 + relu
  gemm16<4, 1, 1><<<dim3(BB / 16, HID / 64), 32, 0, stream>>>(
      hl_h, l0w_h, h1_h, DMODEL, HID, l0_b, bn0_g, bn0_b, bn0_m, bn0_v);

  // 6) lin1 + bn1 + relu
  gemm16<4, 1, 1><<<dim3(BB / 16, HID / 64), 32, 0, stream>>>(
      h1_h, l1w_h, h2_h, HID, HID, l1_b, bn1_g, bn1_b, bn1_m, bn1_v);

  // 7) lin2 + bias + log_softmax
  head_kernel<<<dim3(BB / 16), 32, 0, stream>>>(h2_h, l2w_h, l2_b,
                                                (float*)d_out);
}
